// AttentionBlock_84000970375297
// MI455X (gfx1250) — matrix operand order repaired and checked by reading, not compiled, not run on hardware
//
#include <hip/hip_runtime.h>
#include <hip/hip_bf16.h>

typedef __attribute__((ext_vector_type(16))) _Float16 v16h;
typedef __attribute__((ext_vector_type(8)))  _Float16 v8h;
typedef __attribute__((ext_vector_type(8)))  float    v8f;

#define HEADS 4
#define CCH   256
#define NSP   4096
#define DH    64

// ---------------------------------------------------------------------------
// WMMA helpers (layouts per CDNA5 ISA 7.12.2)
// ---------------------------------------------------------------------------
__device__ __forceinline__ v8f wmma32(v16h a, v16h b, v8f c) {
  return __builtin_amdgcn_wmma_f32_16x16x32_f16(
      /*neg_a=*/false, a, /*neg_b=*/false, b,
      /*c_mod=*/(short)0, c, /*reuse_a=*/false, /*reuse_b=*/false);
}

// A-matrix 16x32 f16 fragment. `row` points at this lane's M-row (row-major,
// K contiguous). Lane (hi = lane>>4): a[0..7]=row[k0+hi*8+i], a[8..15]=row[k0+16+hi*8+i].
__device__ __forceinline__ v16h frag_a(const _Float16* row, int k0, int hi) {
  v8h l = *(const v8h*)(row + k0 + hi * 8);
  v8h h = *(const v8h*)(row + k0 + 16 + hi * 8);
  v16h r;
#pragma unroll
  for (int i = 0; i < 8; ++i) { r[i] = l[i]; r[i + 8] = h[i]; }
  return r;
}

// B-matrix 32x16 f16 fragment. `colk` points at this lane's N-column with the
// 32 K-values stored contiguously. b[0..7]=colk[hi*8+i], b[8..15]=colk[16+hi*8+i].
__device__ __forceinline__ v16h frag_b(const _Float16* colk, int hi) {
  v8h l = *(const v8h*)(colk + hi * 8);
  v8h h = *(const v8h*)(colk + 16 + hi * 8);
  v16h r;
#pragma unroll
  for (int i = 0; i < 8; ++i) { r[i] = l[i]; r[i + 8] = h[i]; }
  return r;
}

// ---------------------------------------------------------------------------
// gfx1250 async global->LDS copy (ASYNCcnt path). Inline asm: portable across
// ROCm 7.2 / amdgpu-toolchain builtin arity differences.
// Each lane copies 16 bytes from global address g into LDS at l.
// ---------------------------------------------------------------------------
__device__ __forceinline__ void async_cp16(const _Float16* g, _Float16* l) {
  unsigned loff =
      (unsigned)(unsigned long long)(__attribute__((address_space(3))) _Float16*)l;
  asm volatile("global_load_async_to_lds_b128 %0, %1, off"
               :: "v"(loff), "v"((unsigned long long)(size_t)g)
               : "memory");
}
__device__ __forceinline__ void wait_async0() {
  asm volatile("s_wait_asynccnt 0x0" ::: "memory");
}

// ---------------------------------------------------------------------------
// Kernel 0: weight fp32 -> f16
// ---------------------------------------------------------------------------
__global__ __launch_bounds__(256) void cvt_w(const float* __restrict__ qw,
                                             const float* __restrict__ pw,
                                             _Float16* __restrict__ qw16,
                                             _Float16* __restrict__ pw16) {
  int i = blockIdx.x * 256 + threadIdx.x;
  if (i < 3 * CCH * CCH) qw16[i] = (_Float16)qw[i];
  if (i < CCH * CCH)     pw16[i] = (_Float16)pw[i];
}

// ---------------------------------------------------------------------------
// Kernel 1: GroupNorm -> h_t (B, N, C) f16 (transposed so GEMM B-frags are
// contiguous in C)
// ---------------------------------------------------------------------------
__global__ __launch_bounds__(256) void gn_kernel(const float* __restrict__ x,
                                                 const float* __restrict__ gw,
                                                 const float* __restrict__ gb,
                                                 _Float16* __restrict__ h_t) {
  const int CG = CCH / 8;                      // 32 channels per group
  int batch = blockIdx.x >> 3;
  int g     = blockIdx.x & 7;
  const float* xg = x + ((size_t)batch * CCH + (size_t)g * CG) * NSP;

  float s = 0.f, ss = 0.f;
  for (int i = threadIdx.x; i < CG * NSP; i += 256) {
    float v = xg[i]; s += v; ss += v * v;
  }
#pragma unroll
  for (int m = 16; m >= 1; m >>= 1) { s += __shfl_xor(s, m); ss += __shfl_xor(ss, m); }
  __shared__ float red[16];
  int wv = threadIdx.x >> 5;
  if ((threadIdx.x & 31) == 0) { red[wv] = s; red[8 + wv] = ss; }
  __syncthreads();
  float S = 0.f, SS = 0.f;
#pragma unroll
  for (int w = 0; w < 8; ++w) { S += red[w]; SS += red[8 + w]; }
  const float invn = 1.0f / (float)(CG * NSP);
  float mean = S * invn;
  float var  = SS * invn - mean * mean;
  float rinv = rsqrtf(var + 1e-5f);

  for (int i = threadIdx.x; i < CG * NSP; i += 256) {
    int cl = i >> 12;                // i / 4096
    int n  = i & (NSP - 1);
    int c  = g * CG + cl;
    float v = (xg[i] - mean) * rinv * gw[c] + gb[c];
    h_t[((size_t)batch * NSP + n) * CCH + c] = (_Float16)v;
  }
}

// ---------------------------------------------------------------------------
// Kernel 2: QKV GEMM (768x4096 = W(768x256) @ h(256x4096)) per batch, WMMA.
// ---------------------------------------------------------------------------
__global__ __launch_bounds__(128) void qkv_gemm(const _Float16* __restrict__ W16,
                                                const float* __restrict__ bias,
                                                const _Float16* __restrict__ h_t,
                                                _Float16* __restrict__ Qm,
                                                _Float16* __restrict__ Km,
                                                _Float16* __restrict__ Vm) {
  int nb = blockIdx.x, mb = blockIdx.y, batch = blockIdx.z;
  int lane = threadIdx.x & 31, wave = threadIdx.x >> 5;
  int hi = lane >> 4, lo = lane & 15;
  int m0 = mb * 64 + (wave >> 1) * 32;
  int n0 = nb * 64 + (wave & 1) * 32;
  const _Float16* hb = h_t + (size_t)batch * NSP * CCH;

  v8f acc[2][2] = {};
  for (int k0 = 0; k0 < CCH; k0 += 32) {
    if (k0 + 32 < CCH)
      __builtin_prefetch(hb + (size_t)(n0 + lo) * CCH + k0 + 32, 0, 1);
    v16h a0 = frag_a(W16 + (size_t)(m0 + lo) * CCH, k0, hi);
    v16h a1 = frag_a(W16 + (size_t)(m0 + 16 + lo) * CCH, k0, hi);
    v16h b0 = frag_b(hb + (size_t)(n0 + lo) * CCH + k0, hi);
    v16h b1 = frag_b(hb + (size_t)(n0 + 16 + lo) * CCH + k0, hi);
    acc[0][0] = wmma32(a0, b0, acc[0][0]);
    acc[0][1] = wmma32(a0, b1, acc[0][1]);
    acc[1][0] = wmma32(a1, b0, acc[1][0]);
    acc[1][1] = wmma32(a1, b1, acc[1][1]);
  }

#pragma unroll
  for (int ti = 0; ti < 2; ++ti)
#pragma unroll
    for (int tj = 0; tj < 2; ++tj)
#pragma unroll
      for (int r = 0; r < 8; ++r) {
        int o = m0 + ti * 16 + r + 8 * hi;     // output channel (0..767)
        int n = n0 + tj * 16 + lo;             // spatial
        _Float16 hv = (_Float16)(acc[ti][tj][r] + bias[o]);
        int part = o >> 8;                     // 0=q 1=k 2=v
        int c = o & 255, hd = c >> 6, dd = c & 63;
        size_t bh = (size_t)batch * HEADS + hd;
        if (part == 0)      Qm[(bh * NSP + n) * DH + dd] = hv;
        else if (part == 1) Km[(bh * NSP + n) * DH + dd] = hv;
        else                Vm[(bh * DH + dd) * NSP + n] = hv;
      }
}

// ---------------------------------------------------------------------------
// Kernel 3: Flash attention with double-buffered async K/V staging in LDS.
// Block = 64 query rows (4 waves x 16-row strip); 64-key tiles; online softmax.
// K tile (64 keys x 64 d) and V tile (64 d x 64 keys, transposed) are loaded
// ONCE per block per tile via global_load_async_to_lds_b128 (ASYNCcnt) while
// the previous tile's WMMAs run.
// ---------------------------------------------------------------------------
__device__ __forceinline__ void load_kv_tile(const _Float16* Kb, const _Float16* Vb,
                                             int kt, _Float16 (*Kt)[64],
                                             _Float16 (*Vt)[64], int tid) {
  // 8KB K + 8KB V, 128 threads * 16B => 4 iterations, 8 async ops / thread.
#pragma unroll
  for (int u = tid; u < 512; u += 128) {
    int row = u >> 3;
    int c   = (u & 7) * 8;
    async_cp16(Kb + (size_t)(kt + row) * DH + c, &Kt[row][c]);
    async_cp16(Vb + (size_t)row * NSP + kt + c, &Vt[row][c]);
  }
}

__global__ __launch_bounds__(128) void flash(const _Float16* __restrict__ Qm,
                                             const _Float16* __restrict__ Km,
                                             const _Float16* __restrict__ Vm,
                                             _Float16* __restrict__ o_t) {
  int qt = blockIdx.x, head = blockIdx.y, batch = blockIdx.z;
  int lane = threadIdx.x & 31, wave = threadIdx.x >> 5;
  int hi = lane >> 4, lo = lane & 15;
  size_t bh = (size_t)batch * HEADS + head;
  const _Float16* Qb = Qm + bh * NSP * DH;
  const _Float16* Kb = Km + bh * NSP * DH;
  const _Float16* Vb = Vm + bh * DH * NSP;
  int qrow0 = qt * 64 + wave * 16;

  v16h qa0 = frag_a(Qb + (size_t)(qrow0 + lo) * DH, 0, hi);
  v16h qa1 = frag_a(Qb + (size_t)(qrow0 + lo) * DH, 32, hi);

  v8f o[4] = {};
  float mrow[8], lrow[8];
#pragma unroll
  for (int r = 0; r < 8; ++r) { mrow[r] = -1e30f; lrow[r] = 0.f; }

  __shared__ _Float16 Ktile[2][64][64];   // (key, d)
  __shared__ _Float16 Vtile[2][64][64];   // (d, key)
  __shared__ _Float16 Pbuf[4][16][64];
  _Float16 (*P)[64] = Pbuf[wave];
  const float scale = 0.125f;  // 1/sqrt(64)

  load_kv_tile(Kb, Vb, 0, Ktile[0], Vtile[0], threadIdx.x);

  for (int kt = 0; kt < NSP; kt += 64) {
    int buf = (kt >> 6) & 1;
    wait_async0();        // this wave's portion of tile `buf` landed
    __syncthreads();      // all waves' portions landed; prev reads of buf^1 done
    if (kt + 64 < NSP)
      load_kv_tile(Kb, Vb, kt + 64, Ktile[buf ^ 1], Vtile[buf ^ 1], threadIdx.x);

    v8f s[4];
#pragma unroll
    for (int j = 0; j < 4; ++j) {
      const _Float16* krow = &Ktile[buf][j * 16 + lo][0];
      v16h kb0 = frag_b(krow, hi);         // d = 0..31
      v16h kb1 = frag_b(krow + 32, hi);    // d = 32..63
      v8f z = {};
      z = wmma32(qa0, kb0, z);
      s[j] = wmma32(qa1, kb1, z);
    }

    // online softmax: per-lane rows r correspond to M = r + 8*hi
#pragma unroll
    for (int r = 0; r < 8; ++r) {
      float mx = -1e30f;
#pragma unroll
      for (int j = 0; j < 4; ++j) { s[j][r] *= scale; mx = fmaxf(mx, s[j][r]); }
#pragma unroll
      for (int m = 8; m >= 1; m >>= 1) mx = fmaxf(mx, __shfl_xor(mx, m));
      float mnew = fmaxf(mrow[r], mx);
      float alpha = __expf(mrow[r] - mnew);
      mrow[r] = mnew;
      float ps = 0.f;
#pragma unroll
      for (int j = 0; j < 4; ++j) {
        float p = __expf(s[j][r] - mnew);
        s[j][r] = p; ps += p;
      }
#pragma unroll
      for (int m = 8; m >= 1; m >>= 1) ps += __shfl_xor(ps, m);
      lrow[r] = alpha * lrow[r] + ps;
#pragma unroll
      for (int dj = 0; dj < 4; ++dj) o[dj][r] *= alpha;
    }

    // stage P (f16) in per-wave LDS, re-read as A-matrix fragments
#pragma unroll
    for (int j = 0; j < 4; ++j)
#pragma unroll
      for (int r = 0; r < 8; ++r)
        P[r + 8 * hi][j * 16 + lo] = (_Float16)s[j][r];

    v16h pa0 = frag_a(&P[lo][0], 0, hi);
    v16h pa1 = frag_a(&P[lo][0], 32, hi);

#pragma unroll
    for (int dj = 0; dj < 4; ++dj) {
      const _Float16* vcol = &Vtile[buf][dj * 16 + lo][0];
      v16h vb0 = frag_b(vcol, hi);         // keys kt..kt+31
      v16h vb1 = frag_b(vcol + 32, hi);    // keys kt+32..kt+63
      o[dj] = wmma32(pa0, vb0, o[dj]);
      o[dj] = wmma32(pa1, vb1, o[dj]);
    }
  }

  float inv[8];
#pragma unroll
  for (int r = 0; r < 8; ++r) inv[r] = 1.0f / lrow[r];
#pragma unroll
  for (int dj = 0; dj < 4; ++dj)
#pragma unroll
    for (int r = 0; r < 8; ++r) {
      int n = qrow0 + r + 8 * hi;
      int c = head * DH + dj * 16 + lo;
      o_t[((size_t)batch * NSP + n) * CCH + c] = (_Float16)(o[dj][r] * inv[r]);
    }
}

// ---------------------------------------------------------------------------
// Kernel 4: Proj GEMM + bias + residual, fp32 out.
// ---------------------------------------------------------------------------
__global__ __launch_bounds__(128) void proj_gemm(const _Float16* __restrict__ W16,
                                                 const float* __restrict__ bias,
                                                 const _Float16* __restrict__ o_t,
                                                 const float* __restrict__ x,
                                                 float* __restrict__ out) {
  int nb = blockIdx.x, mb = blockIdx.y, batch = blockIdx.z;
  int lane = threadIdx.x & 31, wave = threadIdx.x >> 5;
  int hi = lane >> 4, lo = lane & 15;
  int m0 = mb * 64 + (wave >> 1) * 32;
  int n0 = nb * 64 + (wave & 1) * 32;
  const _Float16* ob = o_t + (size_t)batch * NSP * CCH;

  v8f acc[2][2] = {};
  for (int k0 = 0; k0 < CCH; k0 += 32) {
    if (k0 + 32 < CCH)
      __builtin_prefetch(ob + (size_t)(n0 + lo) * CCH + k0 + 32, 0, 1);
    v16h a0 = frag_a(W16 + (size_t)(m0 + lo) * CCH, k0, hi);
    v16h a1 = frag_a(W16 + (size_t)(m0 + 16 + lo) * CCH, k0, hi);
    v16h b0 = frag_b(ob + (size_t)(n0 + lo) * CCH + k0, hi);
    v16h b1 = frag_b(ob + (size_t)(n0 + 16 + lo) * CCH + k0, hi);
    acc[0][0] = wmma32(a0, b0, acc[0][0]);
    acc[0][1] = wmma32(a0, b1, acc[0][1]);
    acc[1][0] = wmma32(a1, b0, acc[1][0]);
    acc[1][1] = wmma32(a1, b1, acc[1][1]);
  }

#pragma unroll
  for (int ti = 0; ti < 2; ++ti)
#pragma unroll
    for (int tj = 0; tj < 2; ++tj)
#pragma unroll
      for (int r = 0; r < 8; ++r) {
        int c = m0 + ti * 16 + r + 8 * hi;
        int n = n0 + tj * 16 + lo;
        size_t idx = ((size_t)batch * CCH + c) * NSP + n;
        out[idx] = x[idx] + bias[c] + acc[ti][tj][r];
      }
}

// ---------------------------------------------------------------------------
extern "C" void kernel_launch(void* const* d_in, const int* in_sizes, int n_in,
                              void* d_out, int out_size, void* d_ws, size_t ws_size,
                              hipStream_t stream) {
  (void)in_sizes; (void)n_in; (void)out_size; (void)ws_size;
  const float* x      = (const float*)d_in[0];
  const float* gn_w   = (const float*)d_in[1];
  const float* gn_b   = (const float*)d_in[2];
  const float* qkv_w  = (const float*)d_in[3];
  const float* qkv_b  = (const float*)d_in[4];
  const float* proj_w = (const float*)d_in[5];
  const float* proj_b = (const float*)d_in[6];
  float* out = (float*)d_out;

  char* ws = (char*)d_ws;
  const size_t MB = (size_t)1 << 20;
  _Float16* h_t  = (_Float16*)(ws + 0 * MB);    // (B,N,C)   8 MB
  _Float16* Qm   = (_Float16*)(ws + 8 * MB);    // (B,h,N,d) 8 MB
  _Float16* Km   = (_Float16*)(ws + 16 * MB);   // (B,h,N,d) 8 MB
  _Float16* Vm   = (_Float16*)(ws + 24 * MB);   // (B,h,d,N) 8 MB
  _Float16* o_t  = (_Float16*)(ws + 32 * MB);   // (B,N,C)   8 MB
  _Float16* qw16 = (_Float16*)(ws + 40 * MB);   // 384 KB
  _Float16* pw16 = (_Float16*)(ws + 40 * MB + 3 * CCH * CCH * sizeof(_Float16));

  cvt_w<<<768, 256, 0, stream>>>(qkv_w, proj_w, qw16, pw16);
  gn_kernel<<<32, 256, 0, stream>>>(x, gn_w, gn_b, h_t);
  qkv_gemm<<<dim3(64, 12, 4), 128, 0, stream>>>(qw16, qkv_b, h_t, Qm, Km, Vm);
  flash<<<dim3(64, HEADS, 4), 128, 0, stream>>>(Qm, Km, Vm, o_t);
  proj_gemm<<<dim3(64, 4, 4), 128, 0, stream>>>(pw16, proj_b, o_t, x, out);
}
